// Tokenizer_83932250898469
// MI455X (gfx1250) — compile-verified, matrix operand order checked by reading, not run on hardware
//
#include <hip/hip_runtime.h>
#include <cmath>

#define BATCH    32
#define NUM_VARS 8
#define EMBED    64
#define SAMPLES  4096
#define WIN      16
#define STEP     4
#define TSTEPS   1020          // len(arange(0, 4096-16, 4))
#define KDIM     1024          // WIN*EMBED
#define MTOT     (BATCH * TSTEPS)   // 32640, multiple of 16

typedef __attribute__((ext_vector_type(16))) _Float16 v16h;
typedef __attribute__((ext_vector_type(8)))  _Float16 v8h;
typedef __attribute__((ext_vector_type(4)))  _Float16 v4h;
typedef __attribute__((ext_vector_type(8)))  float    v8f;

union AFrag { v16h v; v4h q[4]; };
union BFrag { v16h v; v8h h[2]; };

// ---------------------------------------------------------------------------
// Stage 1: e_h[b][d][s] = (f16) floor( sum_v x[b][v][s]*W1[v][d] + b1[d] )
// s fastest -> coalesced loads/stores; (b,d) uniform per 256-thread block.
// ---------------------------------------------------------------------------
__global__ void __launch_bounds__(256)
embed_kernel(const float* __restrict__ x, const float* __restrict__ W1,
             const float* __restrict__ b1, _Float16* __restrict__ eh) {
    int idx = blockIdx.x * blockDim.x + threadIdx.x;   // B*EMBED*SAMPLES threads
    int s  = idx & (SAMPLES - 1);
    int bd = idx >> 12;
    int d  = bd & (EMBED - 1);
    int b  = bd >> 6;
    const float* xb = x + ((size_t)b * NUM_VARS) * SAMPLES + s;
    float acc = b1[d];
#pragma unroll
    for (int v = 0; v < NUM_VARS; ++v)
        acc = fmaf(xb[(size_t)v * SAMPLES], W1[v * EMBED + d], acc);
    eh[idx] = (_Float16)floorf(acc);
}

// ---------------------------------------------------------------------------
// Prep: transpose W2 (1024x64 f32, row-major) -> w2t[n][k] and split into
// f16 hi + f16 residual lo so the WMMA pair (lo then hi) is near-exact f32.
// ---------------------------------------------------------------------------
__global__ void __launch_bounds__(256)
w2prep_kernel(const float* __restrict__ W2, _Float16* __restrict__ hi,
              _Float16* __restrict__ lo) {
    int idx = blockIdx.x * blockDim.x + threadIdx.x;   // EMBED*KDIM threads
    int k = idx & (KDIM - 1);
    int n = idx >> 10;
    float w = W2[(size_t)k * EMBED + n];
    _Float16 h = (_Float16)w;
    hi[idx] = h;
    lo[idx] = (_Float16)(w - (float)h);
}

// ---------------------------------------------------------------------------
// Stage 2 GEMM via V_WMMA_F32_16X16X32_F16.
// M = BATCH*TSTEPS = 32640 (2040 tiles of 16 tokens), N = 64, K = 1024.
// Block = 128 threads = 4 waves; wave w owns output columns [16w, 16w+16).
// K-step kk covers flat features [32kk,32kk+32) == embed dims 2kk,2kk+1 with
// window offsets 0..15 -> contiguous per-lane loads from e_h.
// A-frag (16x32 f16): lane row = lane&15, koff = 8*(lane>>4);
//   elems 0..7  = e_h[b][2kk  ][4t+koff+0..7]
//   elems 8..15 = e_h[b][2kk+1][4t+koff+0..7]
// B-frag (32x16 f16): lane col n = lane&15 (w2t row-major over k);
//   elems 0..7  = w2t[n][32kk+koff+0..7], elems 8..15 = w2t[n][32kk+16+koff+0..7]
// D (16x16 f32): VGPR r -> row 8*(lane>>4)+r, col = lane&15.
// ---------------------------------------------------------------------------
__global__ void __launch_bounds__(128)
gemm_kernel(const _Float16* __restrict__ eh, const _Float16* __restrict__ w2hi,
            const _Float16* __restrict__ w2lo, const float* __restrict__ b2,
            float* __restrict__ out) {
    const int lane  = threadIdx.x & 31;
    const int ntile = threadIdx.x >> 5;            // 0..3
    const int row   = lane & 15;
    const int koff  = (lane >> 4) << 3;            // 0 or 8
    const int m = blockIdx.x * 16 + row;           // token row this lane feeds A from
    const int b = m / TSTEPS;
    const int t = m - b * TSTEPS;

    const _Float16* ap = eh + ((size_t)b * EMBED) * SAMPLES + (size_t)(4 * t + koff);
    const size_t boff  = ((size_t)(ntile * 16 + row)) * KDIM + koff;
    const _Float16* bhp = w2hi + boff;
    const _Float16* blp = w2lo + boff;

    v8f acc = {};
#pragma unroll 4
    for (int kk = 0; kk < KDIM / 32; ++kk) {
        const _Float16* p0 = ap + (size_t)(2 * kk) * SAMPLES;   // dim d = 2kk
        AFrag a;
        a.q[0] = *(const v4h*)(p0);
        a.q[1] = *(const v4h*)(p0 + 4);
        a.q[2] = *(const v4h*)(p0 + SAMPLES);                   // dim d = 2kk+1
        a.q[3] = *(const v4h*)(p0 + SAMPLES + 4);

        BFrag bh, bl;
        bh.h[0] = *(const v8h*)(bhp + kk * 32);
        bh.h[1] = *(const v8h*)(bhp + kk * 32 + 16);
        bl.h[0] = *(const v8h*)(blp + kk * 32);
        bl.h[1] = *(const v8h*)(blp + kk * 32 + 16);

        // residual pass then hi pass, both into the same f32 accumulator
        acc = __builtin_amdgcn_wmma_f32_16x16x32_f16(
            false, a.v, false, bl.v, (short)0, acc, false, false);
        acc = __builtin_amdgcn_wmma_f32_16x16x32_f16(
            false, a.v, false, bh.v, (short)0, acc, false, false);
    }

    const int outcol = ntile * 16 + row;
    const int mbase  = blockIdx.x * 16 + ((lane >> 4) << 3);
    const float bias = b2[outcol];
#pragma unroll
    for (int r = 0; r < 8; ++r)
        out[(size_t)(mbase + r) * EMBED + outcol] = floorf(acc[r] + bias);
}

// ---------------------------------------------------------------------------
extern "C" void kernel_launch(void* const* d_in, const int* in_sizes, int n_in,
                              void* d_out, int out_size, void* d_ws, size_t ws_size,
                              hipStream_t stream) {
    (void)in_sizes; (void)n_in; (void)out_size; (void)ws_size;
    const float* x  = (const float*)d_in[0];   // (32, 8, 4096)
    const float* W1 = (const float*)d_in[1];   // (8, 64)
    const float* b1 = (const float*)d_in[2];   // (64,)
    const float* W2 = (const float*)d_in[3];   // (1024, 64)
    const float* b2 = (const float*)d_in[4];   // (64,)
    float* out = (float*)d_out;                // (32, 1020, 64)

    char* ws = (char*)d_ws;
    _Float16* eh   = (_Float16*)ws;                                      // 16 MB
    _Float16* w2hi = (_Float16*)(ws + (size_t)BATCH * EMBED * SAMPLES * 2);
    _Float16* w2lo = w2hi + (size_t)EMBED * KDIM;                        // 128 KB each

    const int embed_elems = BATCH * EMBED * SAMPLES;      // 8,388,608
    embed_kernel<<<embed_elems / 256, 256, 0, stream>>>(x, W1, b1, eh);

    w2prep_kernel<<<(EMBED * KDIM) / 256, 256, 0, stream>>>(W2, w2hi, w2lo);

    gemm_kernel<<<MTOT / 16, 128, 0, stream>>>(eh, w2hi, w2lo, b2, out);
}
